// GRUlayer_979252544081
// MI455X (gfx1250) — matrix operand order repaired and checked by reading, not compiled, not run on hardware
//
#include <hip/hip_runtime.h>
#include <hip/hip_bf16.h>
#include <stdint.h>

// Problem dims (fixed by the reference)
#define Bq   128
#define Tq   256
#define Iq   512
#define Hq   1024
#define N3H  3072     // 3*H, gate-interleaved columns: n = h*3 + g (g: 0=z,1=i,2=r)
#define NBLK 64       // persistent blocks = N3H / 48

typedef __attribute__((ext_vector_type(16))) __bf16 v16bf;
typedef __attribute__((ext_vector_type(8)))  __bf16 v8bf;
typedef __attribute__((ext_vector_type(8)))  float  v8f;
typedef __attribute__((ext_vector_type(4)))  unsigned v4u;
typedef __attribute__((ext_vector_type(4)))  int      v4i;
typedef __attribute__((ext_vector_type(8)))  int      v8i;

#if defined(__has_builtin)
#if __has_builtin(__builtin_amdgcn_tensor_load_to_lds)
#define USE_TDM 1
#endif
#endif
#ifndef USE_TDM
#define USE_TDM 0
#endif

static __device__ __forceinline__ v16bf join16(v8bf lo, v8bf hi) {
  return __builtin_shufflevector(lo, hi, 0,1,2,3,4,5,6,7,8,9,10,11,12,13,14,15);
}
static __device__ __forceinline__ v8f wmma_bf16(v16bf a, v16bf b, v8f c) {
  // D = A(16x32) * B(32x16) + C(16x16 f32)
  return __builtin_amdgcn_wmma_f32_16x16x32_bf16(false, a, false, b, (short)0, c, false, false);
}

#if USE_TDM
// TDM: 2D tile load, 48 rows x tile_k bf16 elements, from a [N3H][Hq] bf16
// tensor (row stride Hq elements), into LDS at lds_off. Issued once per wave;
// tracked by TENSORcnt. D# layout per CDNA5 ISA ch.8 (group0 128b, group1 256b).
static __device__ __forceinline__ void tdm_load_rows48(unsigned lds_off, const void* gptr,
                                                       unsigned tile_k) {
  unsigned long long ga = (unsigned long long)(uintptr_t)gptr;
  v4u g0 = { 1u,                                   // count=1 (valid), no gather
             lds_off,                              // lds_addr (bytes)
             (unsigned)ga,                         // global_addr[31:0]
             (unsigned)(ga >> 32) | (2u << 30) };  // global_addr[56:32], type=2
  v8i g1 = { 0x00010000,                           // wg_mask=0, data_size=1 (2 bytes)
             (int)(((unsigned)Hq & 0xFFFFu) << 16),  // tensor_dim0[15:0]<<16 (=1024)
             (int)(((unsigned)N3H & 0xFFFFu) << 16), // dim0 hi=0 | tensor_dim1[15:0]<<16
             (int)(tile_k << 16),                    // dim1 hi=0 | tile_dim0<<16
             48,                                     // tile_dim1=48, tile_dim2=0
             Hq,                                     // tensor_dim0_stride (elements)
             0, 0 };                                 // stride0 hi, dim1_stride=0
  v4i g2 = { 0, 0, 0, 0 };
  v4i g3 = { 0, 0, 0, 0 };
#if defined(__clang_major__) && (__clang_major__ >= 23)
  v8i gx = { 0, 0, 0, 0, 0, 0, 0, 0 };
  __builtin_amdgcn_tensor_load_to_lds(g0, g1, g2, g3, gx, 0);
#else
  __builtin_amdgcn_tensor_load_to_lds(g0, g1, g2, g3, 0);
#endif
}
#endif

// ---------------- conversion / packing kernels ----------------

__global__ void cvt_bf16_kernel(__bf16* __restrict__ dst, const float* __restrict__ src, int n) {
  for (int i = blockIdx.x * blockDim.x + threadIdx.x; i < n; i += gridDim.x * blockDim.x)
    dst[i] = (__bf16)src[i];
}

// dst[(h*3+g)*cols + c] = s_g[h*cols + c]   (gate-interleaved, bf16)
__global__ void pack3_kernel(__bf16* __restrict__ dst, const float* __restrict__ s0,
                             const float* __restrict__ s1, const float* __restrict__ s2,
                             int cols, int n) {
  for (int e = blockIdx.x * blockDim.x + threadIdx.x; e < n; e += gridDim.x * blockDim.x) {
    int row = e / cols, c = e - row * cols;
    int h = row / 3, g = row - h * 3;
    const float* s = (g == 0) ? s0 : (g == 1 ? s1 : s2);
    dst[e] = (__bf16)s[h * cols + c];
  }
}

__global__ void packbias_kernel(float* __restrict__ dst, const float* b0, const float* b1,
                                const float* b2, int n3h) {
  int e = blockIdx.x * blockDim.x + threadIdx.x;
  if (e < n3h) { int h = e / 3, g = e - h * 3; dst[e] = (g == 0 ? b0 : (g == 1 ? b1 : b2))[h]; }
}

__global__ void zero2_kernel(uint32_t* a, int na, uint32_t* b, int nb) {
  int i = blockIdx.x * blockDim.x + threadIdx.x;
  if (i < na) a[i] = 0u;
  if (i < nb) b[i] = 0u;
}

// ---------------- Phase 1: feed-forward projections ----------------
// proj[bt][n] = sum_k xbf[bt][k] * Wb[n][k] + biasP[n]   (bf16 WMMA, f32 accum)
// block tile: 128 (M) x 48 (N), K-step 32; 8 waves, wave w owns row-tile w, 3 col-tiles.
__global__ __launch_bounds__(256) void ffproj_kernel(
    const __bf16* __restrict__ xbf, const __bf16* __restrict__ Wb,
    const float* __restrict__ biasP, float* __restrict__ proj) {
  __shared__ __attribute__((aligned(16))) __bf16 Bsh[48 * 32];
  const int tid = threadIdx.x, lane = tid & 31, wave = tid >> 5;
  const int hi = lane >> 4, l15 = lane & 15;
  const int nBase = blockIdx.x * 48;
  const int mBase = blockIdx.y * 128;
  const size_t arowBase = (size_t)(mBase + wave * 16 + l15) * Iq;
  v8f acc0 = {}, acc1 = {}, acc2 = {};
  for (int k0 = 0; k0 < Iq; k0 += 32) {
    if (tid < 192) {                       // stage B tile [48][32] (n-major)
      int row = tid >> 2, q = tid & 3;
      *(uint4*)(&Bsh[row * 32 + q * 8]) =
          *(const uint4*)(&Wb[(size_t)(nBase + row) * Iq + k0 + q * 8]);
    }
    __syncthreads();
    const __bf16* ap = xbf + arowBase + k0; // A frag straight from global (L2-hot)
    v16bf a = join16(*(const v8bf*)(ap + hi * 16), *(const v8bf*)(ap + hi * 16 + 8));
    const __bf16* b0p = &Bsh[(0  + l15) * 32 + hi * 16];
    const __bf16* b1p = &Bsh[(16 + l15) * 32 + hi * 16];
    const __bf16* b2p = &Bsh[(32 + l15) * 32 + hi * 16];
    v16bf b0 = join16(*(const v8bf*)b0p, *(const v8bf*)(b0p + 8));
    v16bf b1 = join16(*(const v8bf*)b1p, *(const v8bf*)(b1p + 8));
    v16bf b2 = join16(*(const v8bf*)b2p, *(const v8bf*)(b2p + 8));
    acc0 = wmma_bf16(a, b0, acc0);
    acc1 = wmma_bf16(a, b1, acc1);
    acc2 = wmma_bf16(a, b2, acc2);
    __syncthreads();
  }
  #pragma unroll
  for (int c = 0; c < 3; ++c) {
    v8f acc = (c == 0) ? acc0 : (c == 1 ? acc1 : acc2);
    int col = nBase + c * 16 + l15;
    float bias = biasP[col];
    #pragma unroll
    for (int r = 0; r < 8; ++r) {
      int m = mBase + wave * 16 + hi * 8 + r;   // C/D layout: m = r + hi*8
      proj[(size_t)m * N3H + col] = acc[r] + bias;
    }
  }
}

// ---------------- Phase 2: persistent recurrent scan ----------------
// 64 blocks, block owns 48 packed columns (16 h-values, all 3 gates).
// The block's entire U panel (48 x 1024 bf16 = 96 KB) is loaded ONCE into LDS by the
// Tensor Data Mover and stays resident for all 256 steps (MI455X: 320 KB LDS/WGP).
// Per step: barrier-free K-loop (global A-frags + LDS B-frags + 3 WMMA per 32-K chunk),
// then in-block gate update. temp/cur/h state lives in registers for the whole scan.
// Device-wide barrier per step via monotonic counter. Dynamic LDS: 96K + 24K = 122880 B.
__global__ __launch_bounds__(256) void recur_kernel(
    const __bf16* __restrict__ Ub, const float* __restrict__ proj,
    const float* __restrict__ ui_b, const float* __restrict__ alpha,
    __bf16* __restrict__ hbuf, float* __restrict__ out, unsigned* __restrict__ syncc) {
  extern __shared__ char smem[];
  __bf16* Bsh  = (__bf16*)smem;                 // [48][1024] resident U panel, 96 KB
  float* accsh = (float*)(smem + 48 * Hq * 2);  // [128][48] gate accums, 24 KB
  const int tid = threadIdx.x, lane = tid & 31, wave = tid >> 5;
  const int hi = lane >> 4, l15 = lane & 15;
  const int nBase = blockIdx.x * 48;
  const int hBase = blockIdx.x * 16;
  const int m0 = wave * 16 + l15;

  // One-shot panel load: Ub rows [nBase, nBase+48), all 1024 k-elements.
#if USE_TDM
  if (tid < 32) {                               // wave 0 drives the Tensor Data Mover
    tdm_load_rows48((unsigned)(uintptr_t)Bsh, Ub + (size_t)nBase * Hq, (unsigned)Hq);
    __builtin_amdgcn_s_wait_tensorcnt(0);
  }
#else
  {
    const uint4* src = (const uint4*)(Ub + (size_t)nBase * Hq);
    uint4* dst = (uint4*)Bsh;
    for (int i = tid; i < 48 * Hq / 8; i += 256) dst[i] = src[i];
  }
#endif

  float tempR[8], curR[8], hR[8], alphR[8], uibR[8];
  #pragma unroll
  for (int j = 0; j < 8; ++j) {
    tempR[j] = 0.f; curR[j] = 0.f; hR[j] = 0.f;
    int hg = hBase + ((tid + j * 256) & 15);
    alphR[j] = alpha[hg];
    uibR[j]  = ui_b[hg];
  }
  __syncthreads();                              // panel visible to all waves

  for (int t = 0; t < Tq; ++t) {
    const __bf16* hin  = hbuf + (size_t)(t & 1) * (Bq * Hq);
    __bf16*       hout = hbuf + (size_t)((t + 1) & 1) * (Bq * Hq);
    {
      int b0i = tid >> 4, h0 = hBase + (tid & 15);
      __builtin_prefetch(&proj[((size_t)b0i * Tq + t) * N3H + (size_t)h0 * 3], 0, 3);
    }
    v8f a0 = {}, a1 = {}, a2 = {};
    #pragma unroll 4
    for (int k0 = 0; k0 < Hq; k0 += 32) {       // barrier-free: B panel is read-only
      const __bf16* ap = hin + (size_t)m0 * Hq + k0;
      v16bf a = join16(*(const v8bf*)(ap + hi * 16), *(const v8bf*)(ap + hi * 16 + 8));
      const __bf16* b0p = &Bsh[(size_t)(0  + l15) * Hq + k0 + hi * 16];
      const __bf16* b1p = &Bsh[(size_t)(16 + l15) * Hq + k0 + hi * 16];
      const __bf16* b2p = &Bsh[(size_t)(32 + l15) * Hq + k0 + hi * 16];
      v16bf bb0 = join16(*(const v8bf*)b0p, *(const v8bf*)(b0p + 8));
      v16bf bb1 = join16(*(const v8bf*)b1p, *(const v8bf*)(b1p + 8));
      v16bf bb2 = join16(*(const v8bf*)b2p, *(const v8bf*)(b2p + 8));
      a0 = wmma_bf16(a, bb0, a0);
      a1 = wmma_bf16(a, bb1, a1);
      a2 = wmma_bf16(a, bb2, a2);
    }
    #pragma unroll
    for (int c = 0; c < 3; ++c) {               // spill gate accums to LDS for per-h gather
      v8f acc = (c == 0) ? a0 : (c == 1 ? a1 : a2);
      #pragma unroll
      for (int r = 0; r < 8; ++r)
        accsh[(wave * 16 + hi * 8 + r) * 48 + c * 16 + l15] = acc[r];
    }
    __syncthreads();
    #pragma unroll
    for (int j = 0; j < 8; ++j) {               // gate update: thread owns (b, hg) pairs
      int e = tid + j * 256;
      int b = e >> 4, hl = e & 15;
      int hg = hBase + hl;
      float az = accsh[b * 48 + hl * 3 + 0];
      float ai = accsh[b * 48 + hl * 3 + 1];
      float ar = accsh[b * 48 + hl * 3 + 2];
      const float* prow = &proj[((size_t)b * Tq + t) * N3H + (size_t)hg * 3];
      float z = 1.f / (1.f + __expf(-(prow[0] + az)));
      float r = 1.f / (1.f + __expf(-(prow[2] + ar)));
      float cur  = alphR[j] * curR[j] + prow[1] + (ai + uibR[j]) * r;
      float temp = z * tempR[j] + (1.f - z) * cur - hR[j];           // Vth = 1
      float h = (temp >= 1.f) ? 1.f : ((temp <= -1.f) ? -1.f : 0.f); // ternary spike
      curR[j] = cur; tempR[j] = temp; hR[j] = h;
      hout[(size_t)b * Hq + hg] = (__bf16)h;
      out[((size_t)b * Tq + t) * Hq + hg] = h;                       // [B,T,H]
    }
    __syncthreads();
    if (tid == 0) {                             // device-wide barrier (monotonic counter)
      __builtin_amdgcn_fence(__ATOMIC_RELEASE, "agent");
      atomicAdd(syncc, 1u);
      unsigned target = (unsigned)(NBLK * (t + 1));
      while (__hip_atomic_load(syncc, __ATOMIC_RELAXED, __HIP_MEMORY_SCOPE_AGENT) < target)
        __builtin_amdgcn_s_sleep(1);
      __builtin_amdgcn_fence(__ATOMIC_ACQUIRE, "agent");
    }
    __syncthreads();
  }
  (void)lane;
}

// ---------------- launch ----------------

extern "C" void kernel_launch(void* const* d_in, const int* in_sizes, int n_in,
                              void* d_out, int out_size, void* d_ws, size_t ws_size,
                              hipStream_t stream) {
  const float* x     = (const float*)d_in[0];
  const float* wz_w  = (const float*)d_in[1];
  const float* wz_b  = (const float*)d_in[2];
  const float* wi_w  = (const float*)d_in[3];
  const float* wi_b  = (const float*)d_in[4];
  const float* wr_w  = (const float*)d_in[5];
  const float* wr_b  = (const float*)d_in[6];
  const float* uz_w  = (const float*)d_in[7];
  const float* ui_w  = (const float*)d_in[8];
  const float* ui_b  = (const float*)d_in[9];
  const float* ur_w  = (const float*)d_in[10];
  const float* alpha = (const float*)d_in[11];
  float* out = (float*)d_out;
  (void)in_sizes; (void)n_in; (void)out_size; (void)ws_size;

  char* ws = (char*)d_ws;
  size_t off = 0;
  auto take = [&](size_t bytes) {
    char* p = ws + off;
    off = (off + bytes + 255) & ~(size_t)255;
    return p;
  };
  __bf16* xbf   = (__bf16*)take((size_t)Bq * Tq * Iq * 2);   // 32 MB
  __bf16* Wb    = (__bf16*)take((size_t)N3H * Iq * 2);       // 3 MB, [3H][I] n-major
  __bf16* Ub    = (__bf16*)take((size_t)N3H * Hq * 2);       // 6 MB, [3H][H] n-major
  float*  biasP = (float*) take((size_t)N3H * 4);
  __bf16* hbuf  = (__bf16*)take((size_t)2 * Bq * Hq * 2);    // ping-pong spikes
  unsigned* syncc = (unsigned*)take(256);
  float*  proj  = (float*) take((size_t)Bq * Tq * N3H * 4);  // 402 MB, [BT][3H]

  int nx = Bq * Tq * Iq;
  cvt_bf16_kernel<<<8192, 256, 0, stream>>>(xbf, x, nx);
  pack3_kernel<<<2048, 256, 0, stream>>>(Wb, wz_w, wi_w, wr_w, Iq, N3H * Iq);
  pack3_kernel<<<4096, 256, 0, stream>>>(Ub, uz_w, ui_w, ur_w, Hq, N3H * Hq);
  packbias_kernel<<<(N3H + 255) / 256, 256, 0, stream>>>(biasP, wz_b, wi_b, wr_b, N3H);
  zero2_kernel<<<(Bq * Hq / 2 + 255) / 256, 256, 0, stream>>>(
      (uint32_t*)hbuf, Bq * Hq / 2, (uint32_t*)syncc, 1);

  dim3 g1(N3H / 48, (Bq * Tq) / 128);   // 64 x 256 blocks
  ffproj_kernel<<<g1, 256, 0, stream>>>(xbf, Wb, biasP, proj);
  // 96 KB resident U panel + 24 KB accumulator staging (MI455X: 320 KB LDS per WGP)
  recur_kernel<<<NBLK, 256, 122880, stream>>>(Ub, proj, ui_b, alpha, hbuf, out, syncc);
}
